// attention_52252572123175
// MI455X (gfx1250) — matrix operand order repaired and checked by reading, not compiled, not run on hardware
//
#include <hip/hip_runtime.h>

// ---------------------------------------------------------------------------
// Types for CDNA5 WMMA (wave32, 16x16x32 bf16 -> f32 accum)
// ---------------------------------------------------------------------------
typedef __attribute__((ext_vector_type(16))) __bf16         v16bf;
typedef __attribute__((ext_vector_type(8)))  float          v8f;
typedef __attribute__((ext_vector_type(8)))  unsigned short v8u16;
typedef __attribute__((ext_vector_type(4)))  unsigned short v4u16;
typedef __attribute__((ext_vector_type(4)))  unsigned int   v4u32;
typedef __attribute__((ext_vector_type(8)))  unsigned int   v8u32;

// Problem constants
#define BATCH 64
#define DDIM  256
#define TDIM  77
#define TPAD  80
#define HWDIM 1024
#define CDIM  768

__device__ __forceinline__ unsigned short f2bf(float x) {
  // round-to-nearest-even f32 -> bf16
  unsigned int u = __float_as_uint(x);
  u += 0x7FFFu + ((u >> 16) & 1u);
  return (unsigned short)(u >> 16);
}

__device__ __forceinline__ v8f vzero8() {
  v8f z = {0.f, 0.f, 0.f, 0.f, 0.f, 0.f, 0.f, 0.f};
  return z;
}

// A-matrix fragment (16x32 bf16, MxK) from row-major [rows][ld] bf16 buffer.
// ISA: lane<16 -> row=lane, K=0..7 & 16..23 ; lane>=16 -> K=8..15 & 24..31.
__device__ __forceinline__ v16bf load_a_frag(const unsigned short* src, int ld) {
  const int lane = (int)(threadIdx.x & 31);
  const unsigned short* p = src + (lane & 15) * ld + ((lane < 16) ? 0 : 8);
  union { v16bf v; v8u16 h[2]; } u;
  u.h[0] = *(const v8u16*)(p);
  u.h[1] = *(const v8u16*)(p + 16);
  return u.v;
}

// B-matrix fragment (32x16 bf16, KxN) from TRANSPOSED storage: row-major
// [N][ld], K contiguous. ISA: lane<16 -> col=lane, K=0..7 & 16..23 ; lane>=16 -> K=8..15 & 24..31.
__device__ __forceinline__ v16bf load_b_frag(const unsigned short* src, int ld) {
  const int lane = (int)(threadIdx.x & 31);
  const unsigned short* p = src + (lane & 15) * ld + ((lane < 16) ? 0 : 8);
  union { v16bf v; v8u16 h[2]; } u;
  u.h[0] = *(const v8u16*)(p);
  u.h[1] = *(const v8u16*)(p + 16);
  return u.v;
}

__device__ __forceinline__ v8f wmma_bf16(v16bf a, v16bf b, v8f c) {
  return __builtin_amdgcn_wmma_f32_16x16x32_bf16(
      false, a, false, b, (short)0, c, false, false);
}

__device__ __forceinline__ void wait_tensorcnt0() {
#if __has_builtin(__builtin_amdgcn_s_wait_tensorcnt)
  __builtin_amdgcn_s_wait_tensorcnt(0);
#else
  asm volatile("s_wait_tensorcnt 0x0" ::: "memory");
#endif
}

// ---------------------------------------------------------------------------
// Prep kernels: one-time f32 -> bf16 conversions (L2-resident results).
// ---------------------------------------------------------------------------
// Wt[d][c] = bf16(W[c][d]) ; grid 768 (c), 256 threads (d)
__global__ __launch_bounds__(256) void k_prep_w(
    const float* __restrict__ W, unsigned short* __restrict__ Wt) {
  const int c = blockIdx.x;
  const int d = (int)threadIdx.x;
  Wt[(size_t)d * CDIM + c] = f2bf(W[(size_t)c * DDIM + d]);
}

// eT[b][t][d] = bf16(e[b][d][t]) with zero padding t in [77,80)
__global__ __launch_bounds__(256) void k_prep_e(
    const float* __restrict__ e, unsigned short* __restrict__ eT) {
  const int t = blockIdx.x;
  const int b = blockIdx.y;
  const int d = (int)threadIdx.x;
  const float val = (t < TDIM) ? e[(size_t)b * (DDIM * TDIM) + (size_t)d * TDIM + t] : 0.f;
  eT[(size_t)b * (TPAD * DDIM) + (size_t)t * DDIM + d] = f2bf(val);
}

// ---------------------------------------------------------------------------
// Kernel 1: v[b][d][n] = sum_c fv[b][n][c] * W[c][d] + bias[d]   (bf16 out)
// Grid (16 n-blocks of 64, BATCH), 256 threads = 8 waves.
// Wave w: rows n = n0 + (w>>1)*16, cols d = (w&1)*128 .. +128 (8 tiles).
// A double-buffered through LDS (f32->bf16 convert overlapped with WMMA);
// all 8 B-frags preloaded per step from global Wt (one clause, one wait).
// ---------------------------------------------------------------------------
__global__ __launch_bounds__(256) void k_proj_v(
    const float* __restrict__ f, const unsigned short* __restrict__ Wt,
    const float* __restrict__ bias, unsigned short* __restrict__ v) {
  const int b   = blockIdx.y;
  const int n0  = blockIdx.x * 64;
  const int tid = (int)threadIdx.x;
  const int w   = tid >> 5;

  __shared__ __align__(16) unsigned short Abuf[2][64 * 32];  // fv tiles, [n][k]

  const float* fv = f + (size_t)b * (HWDIM * CDIM);
  const unsigned short* Wt_w = Wt + (size_t)((w & 1) * 128) * CDIM;

  v8f acc[8];
#pragma unroll
  for (int j = 0; j < 8; ++j) acc[j] = vzero8();

  const int ai = tid >> 2;        // 0..63  A-stage row
  const int ak = (tid & 3) << 3;  // 0,8,16,24
  const float* arow = fv + (size_t)(n0 + ai) * CDIM + ak;

  auto stage = [&](int kk, int sel) {
    const float4* src = (const float4*)(arow + kk);
    float4 x0 = src[0];
    float4 x1 = src[1];
    unsigned short* dst = &Abuf[sel][ai * 32 + ak];
    dst[0] = f2bf(x0.x); dst[1] = f2bf(x0.y);
    dst[2] = f2bf(x0.z); dst[3] = f2bf(x0.w);
    dst[4] = f2bf(x1.x); dst[5] = f2bf(x1.y);
    dst[6] = f2bf(x1.z); dst[7] = f2bf(x1.w);
  };

  stage(0, 0);
  __syncthreads();

  const int NSTEP = CDIM / 32;  // 24
  for (int it = 0; it < NSTEP; ++it) {
    const int cur = it & 1;
    const int kk  = it * 32;
    if (it + 1 < NSTEP) {
      __builtin_prefetch(arow + kk + 64, 0, 1);
      stage(kk + 32, cur ^ 1);  // overlap next A-stage with this step's WMMAs
    }
    v16bf bfr[8];
#pragma unroll
    for (int j = 0; j < 8; ++j)
      bfr[j] = load_b_frag(Wt_w + (size_t)(j * 16) * CDIM + kk, CDIM);
    v16bf a = load_a_frag(&Abuf[cur][(w >> 1) * 16 * 32], 32);
#pragma unroll
    for (int j = 0; j < 8; ++j) acc[j] = wmma_bf16(a, bfr[j], acc[j]);
    __syncthreads();
  }

  // Store: C tile M=n rows, N=d cols. Lane holds fixed d, 8 contiguous n.
  const int lane  = tid & 31;
  const int col   = lane & 15;
  const int rbase = (lane < 16) ? 0 : 8;
  const int nwr   = n0 + (w >> 1) * 16 + rbase;
  unsigned short* vb = v + (size_t)b * (DDIM * HWDIM);
#pragma unroll
  for (int j = 0; j < 8; ++j) {
    const int d  = (w & 1) * 128 + j * 16 + col;
    const float bs = bias[d];
    v8u16 outp;
#pragma unroll
    for (int r = 0; r < 8; ++r) outp[r] = f2bf(acc[j][r] + bs);
    *(v8u16*)(vb + (size_t)d * HWDIM + nwr) = outp;
  }
}

// ---------------------------------------------------------------------------
// Kernel 2: s[b][t][n] = gamma * sum_d e[b][d][t] * v[b][d][n]   (f32 out)
// Grid (8 n-blocks of 128, BATCH), 256 threads = 8 waves.
// e^T tile (80x256 bf16, contiguous) staged into LDS by the Tensor Data
// Mover (tensor_load_to_lds, 2-group D#), completed with s_wait_tensorcnt;
// v tile transposed into LDS cooperatively.
// ---------------------------------------------------------------------------
__global__ __launch_bounds__(256) void k_score(
    const unsigned short* __restrict__ eT, const unsigned short* __restrict__ v,
    const float* __restrict__ gammap, float* __restrict__ s) {
  const int b   = blockIdx.y;
  const int n0  = blockIdx.x * 128;
  const int tid = (int)threadIdx.x;
  const int w   = tid >> 5;
  const float gamma = gammap[0];

  __shared__ __align__(16) unsigned short Ebuf[TPAD * 256];  // e^T [t][d], 40KB
  __shared__ __align__(16) unsigned short Vbuf[128 * 32];    // v^T tile [n][k], 8KB

  // --- TDM: DMA the whole 80x256 bf16 e^T tile into LDS (wave 0 issues) ---
  if (w == 0) {
    const unsigned long long ga =
        (unsigned long long)(const void*)(eT + (size_t)b * (TPAD * DDIM));
    const unsigned lds_addr = (unsigned)(unsigned long long)(const void*)&Ebuf[0];
    v4u32 g0;
    g0[0] = 1u;                                   // count=1, user descriptor
    g0[1] = lds_addr;                             // lds_addr (bytes)
    g0[2] = (unsigned)(ga & 0xFFFFFFFFu);         // global_addr[31:0]
    g0[3] = (unsigned)((ga >> 32) & 0x01FFFFFFu)  // global_addr[56:32]
            | (2u << 30);                         // type = 2 ("image")
    v8u32 g1;
    g1[0] = 0x00010000u;           // workgroup_mask=0, data_size=1 (2 bytes)
    g1[1] = (unsigned)DDIM << 16;  // tensor_dim0 = 256 (bits 79:48 lo)
    g1[2] = (unsigned)TPAD << 16;  // tensor_dim0 hi=0 ; tensor_dim1 = 80 lo
    g1[3] = (unsigned)DDIM << 16;  // tensor_dim1 hi=0 ; tile_dim0 = 256
    g1[4] = (unsigned)TPAD;        // tile_dim1 = 80 ; tile_dim2 = 0
    g1[5] = (unsigned)DDIM;        // tensor_dim0_stride = 256 (lo32)
    g1[6] = 0u;                    // stride0 hi ; tensor_dim1_stride lo
    g1[7] = 0u;
    asm volatile("tensor_load_to_lds %0, %1" :: "s"(g0), "s"(g1) : "memory");
    wait_tensorcnt0();
  }
  __syncthreads();

  v8f acc[5];
#pragma unroll
  for (int j = 0; j < 5; ++j) acc[j] = vzero8();

  const unsigned short* vb = v + (size_t)b * (DDIM * HWDIM);
  for (int kk = 0; kk < DDIM; kk += 32) {
    for (int idx = tid; idx < 128 * 32; idx += 256) {
      const int n = idx & 127, k2 = idx >> 7;
      Vbuf[n * 32 + k2] = vb[(size_t)(kk + k2) * HWDIM + n0 + n];
    }
    __syncthreads();
    v16bf bb = load_b_frag(&Vbuf[w * 16 * 32], 32);
    v16bf afr[5];
#pragma unroll
    for (int j = 0; j < 5; ++j)
      afr[j] = load_a_frag(&Ebuf[j * 16 * 256 + kk], 256);
#pragma unroll
    for (int j = 0; j < 5; ++j) acc[j] = wmma_bf16(afr[j], bb, acc[j]);
    __syncthreads();
  }

  const int lane  = tid & 31;
  const int ncol  = n0 + w * 16 + (lane & 15);
  const int rbase = (lane < 16) ? 0 : 8;
  float* sb = s + (size_t)b * (TPAD * HWDIM);
#pragma unroll
  for (int j = 0; j < 5; ++j)
#pragma unroll
    for (int r = 0; r < 8; ++r) {
      const int t = j * 16 + rbase + r;
      sb[(size_t)t * HWDIM + ncol] = gamma * acc[j][r];
    }
}

// ---------------------------------------------------------------------------
// Kernel 3: alpha[b][t][:] = softmax(s[b][t][:]) over 1024 regions (bf16 out)
// ---------------------------------------------------------------------------
__global__ __launch_bounds__(256) void k_softmax(
    const float* __restrict__ s, unsigned short* __restrict__ alpha) {
  const int b   = blockIdx.x / TPAD;
  const int t   = blockIdx.x % TPAD;
  const int tid = (int)threadIdx.x;
  unsigned short* arow = alpha + (size_t)b * (TPAD * HWDIM) + (size_t)t * HWDIM;
  if (t >= TDIM) {  // zero padded rows so kernel 4 reads clean zeros
    v4u16 z = {0, 0, 0, 0};
    *(v4u16*)(arow + tid * 4) = z;
    return;
  }
  const float* srow = s + (size_t)b * (TPAD * HWDIM) + (size_t)t * HWDIM;
  float4 x = ((const float4*)srow)[tid];

  __shared__ float red[256];
  red[tid] = fmaxf(fmaxf(x.x, x.y), fmaxf(x.z, x.w));
  __syncthreads();
  for (int off = 128; off > 0; off >>= 1) {
    if (tid < off) red[tid] = fmaxf(red[tid], red[tid + off]);
    __syncthreads();
  }
  const float m = red[0];
  __syncthreads();

  const float e0 = __expf(x.x - m), e1 = __expf(x.y - m);
  const float e2 = __expf(x.z - m), e3 = __expf(x.w - m);
  red[tid] = e0 + e1 + e2 + e3;
  __syncthreads();
  for (int off = 128; off > 0; off >>= 1) {
    if (tid < off) red[tid] += red[tid + off];
    __syncthreads();
  }
  const float inv = 1.0f / red[0];
  v4u16 o;
  o[0] = f2bf(e0 * inv); o[1] = f2bf(e1 * inv);
  o[2] = f2bf(e2 * inv); o[3] = f2bf(e3 * inv);
  *(v4u16*)(arow + tid * 4) = o;
}

// ---------------------------------------------------------------------------
// Kernel 4: c[b][d][t] = sum_n v[b][d][n] * alpha[b][t][n]
// Output stored TRANSPOSED: cT[b][t][d] in f32 (norms) and bf16 (ce GEMM).
// LDS-free: both fragments load contiguously from global by layout choice.
// ---------------------------------------------------------------------------
__global__ __launch_bounds__(256) void k_context(
    const unsigned short* __restrict__ v, const unsigned short* __restrict__ alpha,
    float* __restrict__ cTf, unsigned short* __restrict__ cTb) {
  const int b     = blockIdx.y;
  const int tid   = (int)threadIdx.x;
  const int w     = tid >> 5;
  const int dtile = blockIdx.x * 128 + w * 16;

  const unsigned short* vb = v + (size_t)b * (DDIM * HWDIM) + (size_t)dtile * HWDIM;
  const unsigned short* ab = alpha + (size_t)b * (TPAD * HWDIM);

  v8f acc[5];
#pragma unroll
  for (int j = 0; j < 5; ++j) acc[j] = vzero8();

  for (int kk = 0; kk < HWDIM; kk += 32) {
    v16bf a = load_a_frag(vb + kk, HWDIM);  // A: rows d, K=n
    v16bf bfr[5];
#pragma unroll
    for (int j = 0; j < 5; ++j)
      bfr[j] = load_b_frag(ab + (size_t)(j * 16) * HWDIM + kk, HWDIM);
#pragma unroll
    for (int j = 0; j < 5; ++j) acc[j] = wmma_bf16(a, bfr[j], acc[j]);
  }

  const int lane  = tid & 31;
  const int rbase = (lane < 16) ? 0 : 8;
  const int drow  = dtile + rbase;  // 8 contiguous d per lane
#pragma unroll
  for (int j = 0; j < 5; ++j) {
    const int t = j * 16 + (lane & 15);
    float*          dstf = cTf + (size_t)b * (TPAD * DDIM) + (size_t)t * DDIM + drow;
    unsigned short* dstb = cTb + (size_t)b * (TPAD * DDIM) + (size_t)t * DDIM + drow;
    v8u16 ob;
#pragma unroll
    for (int r = 0; r < 8; ++r) { dstf[r] = acc[j][r]; ob[r] = f2bf(acc[j][r]); }
    *(v8u16*)dstb = ob;
  }
}

// ---------------------------------------------------------------------------
// Kernel 5: rle[b][t] = 1/||e[b][:,t]||, rlc[b][t] = 1/||c[b][:,t]||
// ---------------------------------------------------------------------------
__global__ __launch_bounds__(256) void k_norms(
    const float* __restrict__ e, const float* __restrict__ cTf,
    float* __restrict__ rle, float* __restrict__ rlc) {
  const int b   = blockIdx.x / TPAD;
  const int t   = blockIdx.x % TPAD;
  const int tid = (int)threadIdx.x;
  __shared__ float red[256];
  if (t >= TDIM) {
    if (tid == 0) { rle[b * TPAD + t] = 0.f; rlc[b * TPAD + t] = 0.f; }
    return;
  }
  const float ev = e[(size_t)b * (DDIM * TDIM) + (size_t)tid * TDIM + t];
  const float cv = cTf[(size_t)b * (TPAD * DDIM) + (size_t)t * DDIM + tid];

  red[tid] = ev * ev;
  __syncthreads();
  for (int off = 128; off > 0; off >>= 1) {
    if (tid < off) red[tid] += red[tid + off];
    __syncthreads();
  }
  const float se = red[0];
  __syncthreads();

  red[tid] = cv * cv;
  __syncthreads();
  for (int off = 128; off > 0; off >>= 1) {
    if (tid < off) red[tid] += red[tid + off];
    __syncthreads();
  }
  if (tid == 0) {
    rle[b * TPAD + t] = rsqrtf(se);
    rlc[b * TPAD + t] = rsqrtf(red[0]);
  }
}

// ---------------------------------------------------------------------------
// Kernel 6: out[b][t][s] = (sum_d e[b][d][t] * c[b][d][s]) * rle[t] * rlc[s]
// Grid (5 t-tiles, BATCH), 160 threads = 5 waves; wave w owns s-tile w.
// LDS-free: A-frags from pre-converted eT, B-frags from cT (both L2-hot).
// ---------------------------------------------------------------------------
__global__ __launch_bounds__(160) void k_simil(
    const unsigned short* __restrict__ eT, const unsigned short* __restrict__ cTb,
    const float* __restrict__ rle, const float* __restrict__ rlc,
    float* __restrict__ out) {
  const int b   = blockIdx.y;
  const int t0  = blockIdx.x * 16;
  const int tid = (int)threadIdx.x;
  const int w   = tid >> 5;

  const unsigned short* ea = eT + (size_t)b * (TPAD * DDIM) + (size_t)t0 * DDIM;
  const unsigned short* cb = cTb + (size_t)b * (TPAD * DDIM) + (size_t)(w * 16) * DDIM;

  v8f acc = vzero8();
#pragma unroll
  for (int kk = 0; kk < DDIM; kk += 32) {
    v16bf a  = load_a_frag(ea + kk, DDIM);  // eT[t][d]
    v16bf bb = load_b_frag(cb + kk, DDIM);  // cT[s][d]: B[k=d][col=s]
    acc = wmma_bf16(a, bb, acc);
  }

  const int lane  = tid & 31;
  const int scol  = w * 16 + (lane & 15);
  const int rbase = (lane < 16) ? 0 : 8;
  if (scol < TDIM) {
    const float rc = rlc[b * TPAD + scol];
#pragma unroll
    for (int r = 0; r < 8; ++r) {
      const int t = t0 + rbase + r;
      if (t < TDIM)
        out[(size_t)b * (TDIM * TDIM) + (size_t)t * TDIM + scol] =
            acc[r] * rle[b * TPAD + t] * rc;
    }
  }
}

// ---------------------------------------------------------------------------
// Launch: inputs {e, f, gamma, W, b}; output [B,77,77] f32.
// Workspace layout (bytes):
//   v_bf16   @ 0           : 33,554,432
//   s_f32    @ 33,554,432  : 20,971,520
//   alpha    @ 54,525,952  : 10,485,760
//   cT_f32   @ 65,011,712  :  5,242,880
//   cT_bf16  @ 70,254,592  :  2,621,440
//   rle      @ 72,876,032  :     20,480
//   rlc      @ 72,896,512  :     20,480
//   Wt_bf16  @ 72,916,992  :    393,216
//   eT_bf16  @ 73,310,208  :  2,621,440   (total ~76 MB)
// ---------------------------------------------------------------------------
extern "C" void kernel_launch(void* const* d_in, const int* in_sizes, int n_in,
                              void* d_out, int out_size, void* d_ws, size_t ws_size,
                              hipStream_t stream) {
  (void)in_sizes; (void)n_in; (void)out_size; (void)ws_size;
  const float* e     = (const float*)d_in[0];
  const float* f     = (const float*)d_in[1];
  const float* gamma = (const float*)d_in[2];
  const float* W     = (const float*)d_in[3];
  const float* bias  = (const float*)d_in[4];
  float* out = (float*)d_out;

  char* ws = (char*)d_ws;
  unsigned short* v     = (unsigned short*)(ws + 0);
  float*          s     = (float*)         (ws + 33554432);
  unsigned short* alpha = (unsigned short*)(ws + 54525952);
  float*          cTf   = (float*)         (ws + 65011712);
  unsigned short* cTb   = (unsigned short*)(ws + 70254592);
  float*          rle   = (float*)         (ws + 72876032);
  float*          rlc   = (float*)         (ws + 72896512);
  unsigned short* Wt    = (unsigned short*)(ws + 72916992);
  unsigned short* eT    = (unsigned short*)(ws + 73310208);

  k_prep_w <<<dim3(CDIM), 256, 0, stream>>>(W, Wt);
  k_prep_e <<<dim3(TPAD, BATCH), 256, 0, stream>>>(e, eT);
  k_proj_v <<<dim3(16, BATCH), 256, 0, stream>>>(f, Wt, bias, v);
  k_score  <<<dim3(8, BATCH), 256, 0, stream>>>(eT, v, gamma, s);
  k_softmax<<<dim3(BATCH * TPAD), 256, 0, stream>>>(s, alpha);
  k_context<<<dim3(2, BATCH), 256, 0, stream>>>(v, alpha, cTf, cTb);
  k_norms  <<<dim3(BATCH * TPAD), 256, 0, stream>>>(e, cTf, rle, rlc);
  k_simil  <<<dim3(5, BATCH), 160, 0, stream>>>(eT, cTb, rle, rlc, out);
}
